// RelationalSlotAttention_49847390437947
// MI455X (gfx1250) — compile-verified, matrix operand order checked by reading, not run on hardware
//
#include <hip/hip_runtime.h>

// ---------------------------------------------------------------------------
// RelationalSlotAttention on MI455X (gfx1250, wave32, WMMA).
//
// Strategy (compile-only, reasoned from HW specs):
//  * Big GEMMs (LN(x)@Wk^T, LN(x)@Wv^T, q@k^T, attn_n@v) use
//    v_wmma_f32_16x16x32_f16 (f16 in, f32 accumulate). k/v as f16 total
//    67MB -> resident in 192MB L2 across all 3 iterations.
//  * LN is fused into the K/V projection kernel (x read once from HBM).
//  * v is stored transposed [b][d][n] directly from the WMMA D-fragments
//    (D rows are contiguous in n), so the updates GEMM gets clean 128-bit
//    B-fragment loads.
//  * k_lnkv keeps its K loop rolled (#pragma unroll 1): full unroll hoisted
//    32 A-fragments into the extended-VGPR space (s_set_vgpr_msb, ~460
//    VGPRs/wave) which kills occupancy; rolled, the live set is ~150 VGPRs
//    and A-frags re-stream from conflict-free LDS.
//  * GRU/MLP/gates (~1 GFLOP total) are f32 VALU, 8 rows per block to
//    amortize weight streaming from L2. No float atomics -> deterministic.
// ---------------------------------------------------------------------------

typedef _Float16 v16h __attribute__((ext_vector_type(16)));
typedef _Float16 v8h  __attribute__((ext_vector_type(8)));
typedef float    v8f  __attribute__((ext_vector_type(8)));

#define B_    32
#define N_    4096
#define D_    128
#define NS_   8
#define NREL_ 56
#define RTOT_ (B_ * N_)
#define EPS_  1e-8f
#define SCALE_ 0.08838834764831845f   // 128^-0.5

__device__ __forceinline__ float wave_sum(float v) {
#pragma unroll
  for (int m = 16; m >= 1; m >>= 1) v += __shfl_xor(v, m, 32);
  return v;
}

// Assemble a 16-half WMMA fragment from two 128-bit loads.
// A layout (16-bit, 16x32): lane L -> M = L&15; a[j], j<8: K = 8*(L>>4)+j;
// j>=8: K = 16 + 8*(L>>4) + (j-8).  B mirrors with N in place of M.
__device__ __forceinline__ v16h frag16(const _Float16* p0, const _Float16* p1) {
  v8h lo = *(const v8h*)p0;
  v8h hi = *(const v8h*)p1;
  v16h r;
#pragma unroll
  for (int i = 0; i < 8; ++i) { r[i] = lo[i]; r[i + 8] = hi[i]; }
  return r;
}

__device__ __forceinline__ float sigmoidf_(float x) { return 1.0f / (1.0f + expf(-x)); }

// ---------------- weight f16 conversion (Wk, Wv) ----------------
__global__ void k_cvt_w(const float* __restrict__ wk, const float* __restrict__ wv,
                        _Float16* __restrict__ wkh, _Float16* __restrict__ wvh) {
  int i = blockIdx.x * blockDim.x + threadIdx.x;
  if (i < D_ * D_) { wkh[i] = (_Float16)wk[i]; wvh[i] = (_Float16)wv[i]; }
}

// ---------------- slot/rel init: mu + noise*exp(sigma) ----------------
__global__ void k_init(const float* __restrict__ om, const float* __restrict__ os,
                       const float* __restrict__ rm, const float* __restrict__ rs,
                       const float* __restrict__ nobj, const float* __restrict__ nrel,
                       float* __restrict__ obj, float* __restrict__ rel) {
  int i = blockIdx.x * blockDim.x + threadIdx.x;
  if (i < B_ * NS_ * D_)   obj[i] = om[i & 127] + nobj[i] * expf(os[i & 127]);
  if (i < B_ * NREL_ * D_) rel[i] = rm[i & 127] + nrel[i] * expf(rs[i & 127]);
}

// ---------------- fused LN(x) + k = x@Wk^T, v = x@Wv^T (WMMA) ----------------
// Block: 256 thr (8 waves), 128 rows of x. kh row-major [r][d]; vt transposed [b][d][n].
__global__ __launch_bounds__(256)
void k_lnkv(const float* __restrict__ x, const float* __restrict__ lg,
            const float* __restrict__ lb, const _Float16* __restrict__ wkh,
            const _Float16* __restrict__ wvh, _Float16* __restrict__ kh,
            _Float16* __restrict__ vt) {
  const int P = 136;                         // 272B row stride: 16B-aligned, bank-spread
  __shared__ _Float16 xs[128 * P];
  const int row0 = blockIdx.x * 128;
  const int wave = threadIdx.x >> 5, lane = threadIdx.x & 31;
  const float4 gg = *(const float4*)(lg + lane * 4);
  const float4 bb = *(const float4*)(lb + lane * 4);
  for (int t = 0; t < 16; ++t) {             // LN: one wave per row
    const int lr = wave * 16 + t;
    const float4 v = *(const float4*)(x + (size_t)(row0 + lr) * D_ + lane * 4);
    float s = v.x + v.y + v.z + v.w;
    float s2 = v.x * v.x + v.y * v.y + v.z * v.z + v.w * v.w;
    s = wave_sum(s); s2 = wave_sum(s2);
    const float mean = s * (1.0f / 128.0f);
    const float inv = rsqrtf(s2 * (1.0f / 128.0f) - mean * mean + 1e-5f);
    _Float16* d = xs + lr * P + lane * 4;
    d[0] = (_Float16)((v.x - mean) * inv * gg.x + bb.x);
    d[1] = (_Float16)((v.y - mean) * inv * gg.y + bb.y);
    d[2] = (_Float16)((v.z - mean) * inv * gg.z + bb.z);
    d[3] = (_Float16)((v.w - mean) * inv * gg.w + bb.w);
  }
  __syncthreads();
  const int half = lane >> 4, l16 = lane & 15;
  const int nb = wave * 16;                  // 16 output cols per wave
  const int bidx = row0 >> 12;               // batch (128 | 4096)
  const int nrow0 = row0 & (N_ - 1);
  for (int tgt = 0; tgt < 2; ++tgt) {
    const _Float16* W = tgt ? wvh : wkh;
    v8f acc[8];
#pragma unroll
    for (int mt = 0; mt < 8; ++mt) {
#pragma unroll
      for (int j = 0; j < 8; ++j) acc[mt][j] = 0.0f;
    }
#pragma unroll 1                             // keep rolled: VGPR pressure / occupancy
    for (int kk = 0; kk < 4; ++kk) {
      const _Float16* wp = W + (size_t)(nb + l16) * D_ + kk * 32 + half * 8;
      const v16h bf = frag16(wp, wp + 16);   // B(k,n) = W[n][k]
#pragma unroll
      for (int mt = 0; mt < 8; ++mt) {
        const _Float16* ap = xs + (mt * 16 + l16) * P + kk * 32 + half * 8;
        const v16h af = frag16(ap, ap + 16);
        acc[mt] = __builtin_amdgcn_wmma_f32_16x16x32_f16(
            false, af, false, bf, (short)0, acc[mt], false, false);
      }
    }
    if (tgt == 0) {                          // k: row-major f16
#pragma unroll
      for (int mt = 0; mt < 8; ++mt) {
#pragma unroll
        for (int r = 0; r < 8; ++r)
          kh[(size_t)(row0 + mt * 16 + r + half * 8) * D_ + nb + l16] =
              (_Float16)acc[mt][r];
      }
    } else {                                 // v: transposed [b][d][n], packed b128 store
#pragma unroll
      for (int mt = 0; mt < 8; ++mt) {
        v8h pk;
#pragma unroll
        for (int r = 0; r < 8; ++r) pk[r] = (_Float16)acc[mt][r];
        *(v8h*)(vt + ((size_t)bidx * D_ + nb + l16) * N_ + nrow0 + mt * 16 + half * 8) = pk;
      }
    }
  }
}

// ---------------- objn = LN(obj), q = objn @ Wq^T (f32), qh f16 ----------------
__global__ __launch_bounds__(256)
void k_objlnq(const float* __restrict__ obj, const float* __restrict__ lg,
              const float* __restrict__ lb, const float* __restrict__ wq,
              float* __restrict__ objn, _Float16* __restrict__ qh) {
  __shared__ float on[NS_ * D_];
  const int b = blockIdx.x;
  const int wave = threadIdx.x >> 5, lane = threadIdx.x & 31;
  const int row = b * NS_ + wave;
  const float4 v = *(const float4*)(obj + (size_t)row * D_ + lane * 4);
  float s = v.x + v.y + v.z + v.w;
  float s2 = v.x * v.x + v.y * v.y + v.z * v.z + v.w * v.w;
  s = wave_sum(s); s2 = wave_sum(s2);
  const float mean = s * (1.0f / 128.0f);
  const float inv = rsqrtf(s2 * (1.0f / 128.0f) - mean * mean + 1e-5f);
  const float4 gg = *(const float4*)(lg + lane * 4);
  const float4 bb = *(const float4*)(lb + lane * 4);
  float4 o;
  o.x = (v.x - mean) * inv * gg.x + bb.x;
  o.y = (v.y - mean) * inv * gg.y + bb.y;
  o.z = (v.z - mean) * inv * gg.z + bb.z;
  o.w = (v.w - mean) * inv * gg.w + bb.w;
  *(float4*)(objn + (size_t)row * D_ + lane * 4) = o;
  *(float4*)(on + wave * D_ + lane * 4) = o;
  __syncthreads();
  const float* orow = on + wave * D_;
  float acc[4] = {0.f, 0.f, 0.f, 0.f};
  for (int c = 0; c < D_; ++c) {
    const float xv = orow[c];
#pragma unroll
    for (int t = 0; t < 4; ++t) acc[t] += xv * wq[(size_t)(lane * 4 + t) * D_ + c];
  }
  _Float16* qp = qh + (size_t)row * D_ + lane * 4;
#pragma unroll
  for (int t = 0; t < 4; ++t) qp[t] = (_Float16)acc[t];
}

// ---------------- dots = q @ k^T * SCALE  (WMMA) ----------------
__global__ __launch_bounds__(256)
void k_dots(const _Float16* __restrict__ qh, const _Float16* __restrict__ kh,
            float* __restrict__ dots) {
  const int b = blockIdx.y;
  const int wave = threadIdx.x >> 5, lane = threadIdx.x & 31;
  const int half = lane >> 4, l16 = lane & 15;
  const int nbase = blockIdx.x * 128 + wave * 16;
  v8f acc;
#pragma unroll
  for (int j = 0; j < 8; ++j) acc[j] = 0.0f;
#pragma unroll
  for (int kk = 0; kk < 4; ++kk) {
    v16h a;
    if (l16 < NS_) {
      const _Float16* ap = qh + ((size_t)b * NS_ + l16) * D_ + kk * 32 + half * 8;
      a = frag16(ap, ap + 16);
    } else {
#pragma unroll
      for (int j = 0; j < 16; ++j) a[j] = (_Float16)0.0f;
    }
    const _Float16* bp = kh + ((size_t)b * N_ + nbase + l16) * D_ + kk * 32 + half * 8;
    const v16h bf = frag16(bp, bp + 16);
    acc = __builtin_amdgcn_wmma_f32_16x16x32_f16(false, a, false, bf, (short)0, acc,
                                                 false, false);
  }
  if (half == 0) {                           // lanes 0-15 hold rows M=0..7 (all slots)
#pragma unroll
    for (int r = 0; r < NS_; ++r)
      dots[((size_t)b * NS_ + r) * N_ + nbase + l16] = acc[r] * SCALE_;
  }
}

// ---------------- softmax over slots (+EPS) ----------------
__global__ __launch_bounds__(256)
void k_softmax(const float* __restrict__ dots, float* __restrict__ attn) {
  const int b = blockIdx.y;
  const int n = blockIdx.x * 256 + threadIdx.x;
  float d[NS_];
  float mx = -3.4e38f;
#pragma unroll
  for (int s = 0; s < NS_; ++s) { d[s] = dots[((size_t)b * NS_ + s) * N_ + n]; mx = fmaxf(mx, d[s]); }
  float sum = 0.f;
#pragma unroll
  for (int s = 0; s < NS_; ++s) { d[s] = expf(d[s] - mx); sum += d[s]; }
  const float inv = 1.0f / sum;
#pragma unroll
  for (int s = 0; s < NS_; ++s) attn[((size_t)b * NS_ + s) * N_ + n] = d[s] * inv + EPS_;
}

// ---------------- deterministic row sums of attn ----------------
__global__ __launch_bounds__(256)
void k_asum(const float* __restrict__ attn, float* __restrict__ asum) {
  __shared__ float part[8];
  const int row = blockIdx.x;                // 0..255 = (b, s)
  const int t = threadIdx.x;
  float s = 0.f;
  for (int k = 0; k < 16; ++k) s += attn[(size_t)row * N_ + k * 256 + t];
  s = wave_sum(s);
  if ((t & 31) == 0) part[t >> 5] = s;
  __syncthreads();
  if (t == 0) {
    float tot = 0.f;
    for (int w = 0; w < 8; ++w) tot += part[w];
    asum[row] = tot;
  }
}

// ---------------- attn_n = attn / rowsum, as f16 ----------------
__global__ __launch_bounds__(256)
void k_attnnorm(const float* __restrict__ attn, const float* __restrict__ asum,
                _Float16* __restrict__ anh) {
  const int b = blockIdx.y;
  const int n = blockIdx.x * 256 + threadIdx.x;
#pragma unroll
  for (int s = 0; s < NS_; ++s)
    anh[((size_t)b * NS_ + s) * N_ + n] =
        (_Float16)(attn[((size_t)b * NS_ + s) * N_ + n] / asum[b * NS_ + s]);
}

// ---------------- updates = attn_n @ v  (WMMA, K=4096) ----------------
__global__ __launch_bounds__(256)
void k_updates(const _Float16* __restrict__ anh, const _Float16* __restrict__ vt,
               float* __restrict__ upd) {
  const int b = blockIdx.x;
  const int wave = threadIdx.x >> 5, lane = threadIdx.x & 31;
  const int half = lane >> 4, l16 = lane & 15;
  const int dbase = wave * 16;
  v8f acc;
#pragma unroll
  for (int j = 0; j < 8; ++j) acc[j] = 0.0f;
  for (int kk = 0; kk < N_ / 32; ++kk) {
    v16h a;
    if (l16 < NS_) {
      const _Float16* ap = anh + ((size_t)b * NS_ + l16) * N_ + kk * 32 + half * 8;
      a = frag16(ap, ap + 16);
    } else {
#pragma unroll
      for (int j = 0; j < 16; ++j) a[j] = (_Float16)0.0f;
    }
    const _Float16* bp = vt + ((size_t)b * D_ + dbase + l16) * N_ + kk * 32 + half * 8;
    const v16h bf = frag16(bp, bp + 16);
    acc = __builtin_amdgcn_wmma_f32_16x16x32_f16(false, a, false, bf, (short)0, acc,
                                                 false, false);
  }
  if (half == 0) {
#pragma unroll
    for (int r = 0; r < NS_; ++r)
      upd[((size_t)b * NS_ + r) * D_ + dbase + l16] = acc[r];
  }
}

// ---------------- relational gates gi/gj per off-diagonal pair ----------------
__global__ __launch_bounds__(256)
void k_gates(const float* __restrict__ objn, const float* __restrict__ rel,
             const float* __restrict__ wsg, const float* __restrict__ wsb,
             const float* __restrict__ wog, const float* __restrict__ wob,
             float* __restrict__ gi, float* __restrict__ gj) {
  const int b = blockIdx.x;
  const int wave = threadIdx.x >> 5, lane = threadIdx.x & 31;
  const float4 s0 = *(const float4*)(wsg + lane * 4);
  const float4 s1 = *(const float4*)(wsg + 128 + lane * 4);
  const float4 o0 = *(const float4*)(wog + lane * 4);
  const float4 o1 = *(const float4*)(wog + 128 + lane * 4);
  for (int t = 0; t < 7; ++t) {
    const int e = wave * 7 + t;              // 0..55
    const int i = e / 7, q = e % 7;
    const int j = q + (q >= i ? 1 : 0);
    const float4 xi = *(const float4*)(objn + ((size_t)b * NS_ + i) * D_ + lane * 4);
    const float4 xj = *(const float4*)(objn + ((size_t)b * NS_ + j) * D_ + lane * 4);
    const float4 rr = *(const float4*)(rel + ((size_t)b * NREL_ + e) * D_ + lane * 4);
    float si = xi.x * s0.x + xi.y * s0.y + xi.z * s0.z + xi.w * s0.w +
               rr.x * s1.x + rr.y * s1.y + rr.z * s1.z + rr.w * s1.w;
    float sj = xj.x * o0.x + xj.y * o0.y + xj.z * o0.z + xj.w * o0.w +
               rr.x * o1.x + rr.y * o1.y + rr.z * o1.z + rr.w * o1.w;
    si = wave_sum(si); sj = wave_sum(sj);
    if (lane == 0) {
      gi[(size_t)b * NREL_ + e] = sigmoidf_(si + wsb[0]);
      gj[(size_t)b * NREL_ + e] = sigmoidf_(sj + wob[0]);
    }
  }
}

// ---------------- ctx[b,s] = (sum_j gi*rel + sum_j gj*rel)/14 ----------------
__global__ __launch_bounds__(128)
void k_ctx(const float* __restrict__ gi, const float* __restrict__ gj,
           const float* __restrict__ rel, float* __restrict__ ctx) {
  const int b = blockIdx.y, s = blockIdx.x, c = threadIdx.x;
  float acc = 0.f;
  for (int q = 0; q < 7; ++q) {
    const int j = q + (q >= s ? 1 : 0);
    const int e1 = 7 * s + q;                       // pair (s, j)
    const int e2 = 7 * j + s - (s > j ? 1 : 0);     // pair (j, s)
    acc += gi[(size_t)b * NREL_ + e1] * rel[((size_t)b * NREL_ + e1) * D_ + c];
    acc += gj[(size_t)b * NREL_ + e2] * rel[((size_t)b * NREL_ + e2) * D_ + c];
  }
  ctx[((size_t)b * NS_ + s) * D_ + c] = acc * (1.0f / 14.0f);
}

// ---------------- fused GRU + residual MLP, 8 rows/block, in-place ----------------
__global__ __launch_bounds__(128)
void k_gru(int mode, const float* __restrict__ upd, const float* __restrict__ ctx,
           float* __restrict__ obj, float* __restrict__ rel,
           const float* __restrict__ wih, const float* __restrict__ whh,
           const float* __restrict__ bih, const float* __restrict__ bhh,
           const float* __restrict__ lng, const float* __restrict__ lnb,
           const float* __restrict__ l1w, const float* __restrict__ l1b,
           const float* __restrict__ l2w, const float* __restrict__ l2b) {
  __shared__ float sin_[8][2 * D_];
  __shared__ float sh_[8][D_];
  __shared__ float sx_[8][D_];
  __shared__ float partS[4], partS2[4];
  const int t = threadIdx.x;
  const int row0 = blockIdx.x * 8;
  for (int rr = 0; rr < 8; ++rr) {
    const int row = row0 + rr;
    if (mode == 0) {
      sin_[rr][t]       = upd[(size_t)row * D_ + t];
      sin_[rr][t + D_]  = ctx[(size_t)row * D_ + t];
      sh_[rr][t]        = obj[(size_t)row * D_ + t];
    } else {
      const int b = row / NREL_, e = row % NREL_;
      const int i = e / 7, q = e % 7;
      const int j = q + (q >= i ? 1 : 0);
      sin_[rr][t]       = obj[((size_t)b * NS_ + i) * D_ + t];
      sin_[rr][t + D_]  = obj[((size_t)b * NS_ + j) * D_ + t];
      sh_[rr][t]        = rel[(size_t)row * D_ + t];
    }
  }
  __syncthreads();
  // gates: thread t owns gate components t (r), t+128 (z), t+256 (g)
  float g1[8] = {0}, g2[8] = {0}, g3[8] = {0};
  {
    const float* w1 = wih + (size_t)t * 2 * D_;
    const float* w2 = wih + (size_t)(t + D_) * 2 * D_;
    const float* w3 = wih + (size_t)(t + 2 * D_) * 2 * D_;
    for (int c = 0; c < 2 * D_; ++c) {
      const float a = w1[c], b = w2[c], cc = w3[c];
#pragma unroll
      for (int rr = 0; rr < 8; ++rr) {
        const float xv = sin_[rr][c];
        g1[rr] += a * xv; g2[rr] += b * xv; g3[rr] += cc * xv;
      }
    }
  }
  float h1[8] = {0}, h2[8] = {0}, h3[8] = {0};
  {
    const float* v1 = whh + (size_t)t * D_;
    const float* v2 = whh + (size_t)(t + D_) * D_;
    const float* v3 = whh + (size_t)(t + 2 * D_) * D_;
    for (int c = 0; c < D_; ++c) {
      const float a = v1[c], b = v2[c], cc = v3[c];
#pragma unroll
      for (int rr = 0; rr < 8; ++rr) {
        const float hv = sh_[rr][c];
        h1[rr] += a * hv; h2[rr] += b * hv; h3[rr] += cc * hv;
      }
    }
  }
  const float bi1 = bih[t], bi2 = bih[t + D_], bi3 = bih[t + 2 * D_];
  const float bh1 = bhh[t], bh2 = bhh[t + D_], bh3 = bhh[t + 2 * D_];
  float newh[8];
#pragma unroll
  for (int rr = 0; rr < 8; ++rr) {
    const float r_ = sigmoidf_(g1[rr] + bi1 + h1[rr] + bh1);
    const float z_ = sigmoidf_(g2[rr] + bi2 + h2[rr] + bh2);
    const float gg = tanhf(g3[rr] + bi3 + r_ * (h3[rr] + bh3));
    newh[rr] = (1.0f - z_) * gg + z_ * sh_[rr][t];
  }
  __syncthreads();
  for (int rr = 0; rr < 8; ++rr) sh_[rr][t] = newh[rr];
  __syncthreads();
  // LN per row -> sx_
  for (int rr = 0; rr < 8; ++rr) {
    const float xv = sh_[rr][t];
    float s = wave_sum(xv), s2 = wave_sum(xv * xv);
    if ((t & 31) == 0) { partS[t >> 5] = s; partS2[t >> 5] = s2; }
    __syncthreads();
    const float S = partS[0] + partS[1] + partS[2] + partS[3];
    const float S2 = partS2[0] + partS2[1] + partS2[2] + partS2[3];
    const float mean = S * (1.0f / 128.0f);
    const float inv = rsqrtf(S2 * (1.0f / 128.0f) - mean * mean + 1e-5f);
    sx_[rr][t] = (xv - mean) * inv * lng[t] + lnb[t];
    __syncthreads();
  }
  // hid = relu(l1(sx))
  float ha[8] = {0};
  {
    const float* wr = l1w + (size_t)t * D_;
    for (int c = 0; c < D_; ++c) {
      const float w = wr[c];
#pragma unroll
      for (int rr = 0; rr < 8; ++rr) ha[rr] += sx_[rr][c] * w;
    }
  }
  const float b1 = l1b[t];
  __syncthreads();
  for (int rr = 0; rr < 8; ++rr) sx_[rr][t] = fmaxf(ha[rr] + b1, 0.0f);
  __syncthreads();
  // out = newh + l2(hid)
  float oa[8] = {0};
  {
    const float* wr = l2w + (size_t)t * D_;
    for (int c = 0; c < D_; ++c) {
      const float w = wr[c];
#pragma unroll
      for (int rr = 0; rr < 8; ++rr) oa[rr] += sx_[rr][c] * w;
    }
  }
  const float b2 = l2b[t];
  float* outp = (mode == 0) ? obj : rel;
#pragma unroll
  for (int rr = 0; rr < 8; ++rr)
    outp[(size_t)(row0 + rr) * D_ + t] = newh[rr] + oa[rr] + b2;
}

// ---------------- final slots = concat(obj, rel) along axis 1 ----------------
__global__ void k_slots(const float* __restrict__ obj, const float* __restrict__ rel,
                        float* __restrict__ out) {
  const int idx = blockIdx.x * 256 + threadIdx.x;
  if (idx >= B_ * 64 * D_) return;
  const int c = idx & 127;
  const int r = (idx >> 7) & 63;
  const int b = idx >> 13;
  out[idx] = (r < NS_) ? obj[((size_t)b * NS_ + r) * D_ + c]
                       : rel[((size_t)b * NREL_ + (r - NS_)) * D_ + c];
}

// ---------------------------------------------------------------------------
// d_in is jax tree_leaves order (dicts sorted by key):
//  0 inputs  1 noise_obj  2 noise_rel  3 Wk  4 Wq  5 Wv
//  6/7 norm_input.b/g  8/9 norm_obj_slots.b/g  10/11 norm_pre_ff_obj.b/g
//  12/13 norm_pre_ff_rel.b/g
//  14-17 obj_gru.{bhh,bih,whh,wih}  18-21 obj_mlp.{l1.b,l1.w,l2.b,l2.w}
//  22 obj_mu  23 obj_sigma
//  24-27 rel_gru.{bhh,bih,whh,wih}  28-31 rel_mlp.{l1.b,l1.w,l2.b,l2.w}
//  32 rel_mu  33/34 rel_o_gate.b/w  35/36 rel_s_gate.b/w  37 rel_sigma
// d_out: [slots 32*64*128][attn0 32*8*4096][attn1][attn2]  (f32)
// Workspace use: ~75 MB.
// ---------------------------------------------------------------------------
extern "C" void kernel_launch(void* const* d_in, const int* in_sizes, int n_in,
                              void* d_out, int out_size, void* d_ws, size_t ws_size,
                              hipStream_t stream) {
  (void)in_sizes; (void)n_in; (void)out_size; (void)ws_size;
  const float* in_x   = (const float*)d_in[0];
  const float* nobj   = (const float*)d_in[1];
  const float* nrel   = (const float*)d_in[2];
  const float* Wk     = (const float*)d_in[3];
  const float* Wq     = (const float*)d_in[4];
  const float* Wv     = (const float*)d_in[5];
  const float* ni_b   = (const float*)d_in[6];
  const float* ni_g   = (const float*)d_in[7];
  const float* nos_b  = (const float*)d_in[8];
  const float* nos_g  = (const float*)d_in[9];
  const float* npo_b  = (const float*)d_in[10];
  const float* npo_g  = (const float*)d_in[11];
  const float* npr_b  = (const float*)d_in[12];
  const float* npr_g  = (const float*)d_in[13];
  const float* og_bhh = (const float*)d_in[14];
  const float* og_bih = (const float*)d_in[15];
  const float* og_whh = (const float*)d_in[16];
  const float* og_wih = (const float*)d_in[17];
  const float* ol1_b  = (const float*)d_in[18];
  const float* ol1_w  = (const float*)d_in[19];
  const float* ol2_b  = (const float*)d_in[20];
  const float* ol2_w  = (const float*)d_in[21];
  const float* obj_mu = (const float*)d_in[22];
  const float* obj_sg = (const float*)d_in[23];
  const float* rg_bhh = (const float*)d_in[24];
  const float* rg_bih = (const float*)d_in[25];
  const float* rg_whh = (const float*)d_in[26];
  const float* rg_wih = (const float*)d_in[27];
  const float* rl1_b  = (const float*)d_in[28];
  const float* rl1_w  = (const float*)d_in[29];
  const float* rl2_b  = (const float*)d_in[30];
  const float* rl2_w  = (const float*)d_in[31];
  const float* rel_mu = (const float*)d_in[32];
  const float* rog_b  = (const float*)d_in[33];
  const float* rog_w  = (const float*)d_in[34];
  const float* rsg_b  = (const float*)d_in[35];
  const float* rsg_w  = (const float*)d_in[36];
  const float* rel_sg = (const float*)d_in[37];

  size_t off = 0;
  auto alloc = [&](size_t bytes) -> void* {
    void* p = (char*)d_ws + off;
    off += (bytes + 255) & ~(size_t)255;
    return p;
  };
  _Float16* kh   = (_Float16*)alloc((size_t)RTOT_ * D_ * 2);
  _Float16* vt   = (_Float16*)alloc((size_t)RTOT_ * D_ * 2);
  _Float16* wkh  = (_Float16*)alloc((size_t)D_ * D_ * 2);
  _Float16* wvh  = (_Float16*)alloc((size_t)D_ * D_ * 2);
  float*    obj  = (float*)alloc((size_t)B_ * NS_ * D_ * 4);
  float*    rel  = (float*)alloc((size_t)B_ * NREL_ * D_ * 4);
  float*    objn = (float*)alloc((size_t)B_ * NS_ * D_ * 4);
  _Float16* qh   = (_Float16*)alloc((size_t)B_ * NS_ * D_ * 2);
  float*    dots = (float*)alloc((size_t)B_ * NS_ * N_ * 4);
  float*    asum = (float*)alloc((size_t)B_ * NS_ * 4);
  _Float16* anh  = (_Float16*)alloc((size_t)B_ * NS_ * N_ * 2);
  float*    upd  = (float*)alloc((size_t)B_ * NS_ * D_ * 4);
  float*    ctx  = (float*)alloc((size_t)B_ * NS_ * D_ * 4);
  float*    gi   = (float*)alloc((size_t)B_ * NREL_ * 4);
  float*    gj   = (float*)alloc((size_t)B_ * NREL_ * 4);

  float* out = (float*)d_out;
  float* attn_base = out + (size_t)B_ * 64 * D_;

  k_cvt_w<<<64, 256, 0, stream>>>(Wk, Wv, wkh, wvh);
  k_init<<<(B_ * NREL_ * D_ + 255) / 256, 256, 0, stream>>>(
      obj_mu, obj_sg, rel_mu, rel_sg, nobj, nrel, obj, rel);
  k_lnkv<<<RTOT_ / 128, 256, 0, stream>>>(in_x, ni_g, ni_b, wkh, wvh, kh, vt);

  for (int it = 0; it < 3; ++it) {
    float* attn = attn_base + (size_t)it * B_ * NS_ * N_;
    k_objlnq<<<B_, 256, 0, stream>>>(obj, nos_g, nos_b, Wq, objn, qh);
    k_dots<<<dim3(N_ / 128, B_), 256, 0, stream>>>(qh, kh, dots);
    k_softmax<<<dim3(N_ / 256, B_), 256, 0, stream>>>(dots, attn);
    k_asum<<<B_ * NS_, 256, 0, stream>>>(attn, asum);
    k_attnnorm<<<dim3(N_ / 256, B_), 256, 0, stream>>>(attn, asum, anh);
    k_updates<<<B_, 256, 0, stream>>>(anh, vt, upd);
    k_gates<<<B_, 256, 0, stream>>>(objn, rel, rsg_w, rsg_b, rog_w, rog_b, gi, gj);
    k_ctx<<<dim3(NS_, B_), 128, 0, stream>>>(gi, gj, rel, ctx);
    k_gru<<<(B_ * NS_) / 8, 128, 0, stream>>>(0, upd, ctx, obj, rel, og_wih, og_whh,
                                              og_bih, og_bhh, npo_g, npo_b, ol1_w,
                                              ol1_b, ol2_w, ol2_b);
    k_gru<<<(B_ * NREL_) / 8, 128, 0, stream>>>(1, upd, ctx, obj, rel, rg_wih, rg_whh,
                                                rg_bih, rg_bhh, npr_g, npr_b, rl1_w,
                                                rl1_b, rl2_w, rl2_b);
  }
  k_slots<<<(B_ * 64 * D_) / 256, 256, 0, stream>>>(obj, rel, out);
}
